// StackSAModuleAttMSG_51015621542423
// MI455X (gfx1250) — matrix operand order repaired and checked by reading, not compiled, not run on hardware
//
#include <hip/hip_runtime.h>
#include <hip/hip_bf16.h>

typedef _Float16 v16h __attribute__((ext_vector_type(16)));
typedef float    v8f  __attribute__((ext_vector_type(8)));

#define IN_CHN  8
#define OUT_CHN 16
#define HIDDEN  64
#define NS      32
#define WPB     8          // waves per block (256 threads)
#define RADIUS_INV 1.25f
#define EPS 1e-5f

// ---------------- workspace layout (floats) ----------------
// [0..4]     S1 (sum g5)            (5)
// [8..32]    S2 (sum g5 g5^T)       (25)
// [40..103]  scale1[64]
// [104..167] bias1[64]
// [168..183] sum2[16]
// [184..199] sumsq2[16]
// [256.. )   pooled[M*16]

__global__ __launch_bounds__(256)
void zero_kernel(float* p, int n) {
    for (int i = blockIdx.x * blockDim.x + threadIdx.x; i < n; i += gridDim.x * blockDim.x)
        p[i] = 0.0f;
}

// ---- pass 1: first/second moments of g5 over all L rows (BN1 via moment trick:
// mean(g5@W1^T) = mean(g5)@W1^T ; E[(g5.w)^2] = w^T E[g5 g5^T] w )
__global__ __launch_bounds__(256)
void stats1_kernel(const float* __restrict__ xyz, const float* __restrict__ new_xyz,
                   const int* __restrict__ group_idx, float* __restrict__ ws, int M) {
    __shared__ float acc[32];
    int tid = threadIdx.x, w = tid >> 5, lane = tid & 31;
    if (tid < 32) acc[tid] = 0.0f;
    __syncthreads();
    int m = blockIdx.x * WPB + w;
    if (m < M) {
        int idx = group_idx[m * NS + lane];
        float rx = xyz[idx * 3 + 0] - new_xyz[m * 3 + 0];
        float ry = xyz[idx * 3 + 1] - new_xyz[m * 3 + 1];
        float rz = xyz[idx * 3 + 2] - new_xyz[m * 3 + 2];
        float d  = sqrtf(rx * rx + ry * ry + rz * rz);
        float g[5] = { rx, ry, rz, d, d * RADIUS_INV };
        #pragma unroll
        for (int j = 0; j < 5; j++) atomicAdd(&acc[j], g[j]);          // ds_add_f32
        #pragma unroll
        for (int j = 0; j < 5; j++)
            #pragma unroll
            for (int k = 0; k < 5; k++) atomicAdd(&acc[5 + j * 5 + k], g[j] * g[k]);
    }
    __syncthreads();
    if (tid < 30) atomicAdd(&ws[tid < 5 ? tid : 8 + (tid - 5)], acc[tid]);
}

// ---- BN1 finalize: per-channel scale/bias from moments. h = relu(hpre*scale + bias)
__global__ void bn1_finalize_kernel(const float* __restrict__ W1,
                                    const float* __restrict__ gamma, const float* __restrict__ beta,
                                    float* __restrict__ ws, float invL) {
    int c = threadIdx.x;            // 64 threads
    float wv[5];
    #pragma unroll
    for (int j = 0; j < 5; j++) wv[j] = W1[c * 5 + j];
    float mu = 0.0f;
    #pragma unroll
    for (int j = 0; j < 5; j++) mu += ws[j] * wv[j];
    mu *= invL;
    float ex2 = 0.0f;
    #pragma unroll
    for (int j = 0; j < 5; j++)
        #pragma unroll
        for (int k = 0; k < 5; k++) ex2 += ws[8 + j * 5 + k] * wv[j] * wv[k];
    ex2 *= invL;
    float var = ex2 - mu * mu;
    float rs  = rsqrtf(fmaxf(var, 0.0f) + EPS);
    float sc  = rs * gamma[c];
    ws[40 + c]  = sc;
    ws[104 + c] = beta[c] - mu * sc;
}

// ---- fused main pass: g5 -> h(f16, LDS) -> WMMA f16 GEMM vs W2 -> in-register
// attention pooling -> pooled (M x 16). One wave per keypoint m.
__global__ __launch_bounds__(256)
void attn_wmma_kernel(const float* __restrict__ xyz, const float* __restrict__ new_xyz,
                      const float* __restrict__ features, const int* __restrict__ group_idx,
                      const float* __restrict__ W1, const float* __restrict__ W2,
                      const float* __restrict__ scale1, const float* __restrict__ bias1,
                      float* __restrict__ pooled, int M) {
    __shared__ _Float16 sW2[128 * 64];            // 16 KB: B operand, row n = output chan o*8+i
    __shared__ float    sW1[64 * 5];
    __shared__ float    sScale[64], sBias[64];
    __shared__ _Float16 sH[WPB][NS][64];          // 32 KB: A operand tiles (per wave)
    __shared__ float    sFeat[WPB][NS][IN_CHN];   // 8 KB
    __shared__ float    sOut[WPB][OUT_CHN];

    int tid = threadIdx.x;
    for (int i = tid; i < 128 * 64; i += 256) sW2[i] = (_Float16)W2[i];
    for (int i = tid; i < 64 * 5;   i += 256) sW1[i] = W1[i];
    if (tid < 64) { sScale[tid] = scale1[tid]; sBias[tid] = bias1[tid]; }

    int w = tid >> 5, lane = tid & 31;
    int m = blockIdx.x * WPB + w;
    __syncthreads();

    if (m < M) {
        int idx = group_idx[m * NS + lane];
        float rx = xyz[idx * 3 + 0] - new_xyz[m * 3 + 0];
        float ry = xyz[idx * 3 + 1] - new_xyz[m * 3 + 1];
        float rz = xyz[idx * 3 + 2] - new_xyz[m * 3 + 2];
        float d  = sqrtf(rx * rx + ry * ry + rz * rz);
        float g[5] = { rx, ry, rz, d, d * RADIUS_INV };
        #pragma unroll
        for (int i = 0; i < IN_CHN; i++) sFeat[w][lane][i] = features[idx * IN_CHN + i];
        // h row: Linear(5->64) + BN1 + ReLU, stored f16
        for (int c = 0; c < HIDDEN; c++) {
            float a = 0.0f;
            #pragma unroll
            for (int j = 0; j < 5; j++) a = fmaf(g[j], sW1[c * 5 + j], a);
            a = fmaxf(fmaf(a, sScale[c], sBias[c]), 0.0f);
            sH[w][lane][c] = (_Float16)a;
        }
        if (lane < OUT_CHN) sOut[w][lane] = 0.0f;
    }
    __syncthreads();

    if (m < M) {
        int khalf = lane >> 4;        // K half selector per ISA layout
        int hh    = lane & 15;        // row (A) / column (B,C,D)
        int i     = hh & 7;           // input channel this lane consumes

        // ---- hoist A fragments: rows hh (rt=0) and 16+hh (rt=1), K halves 0..31 / 32..63
        v16h a00, a01, a10, a11;
        #pragma unroll
        for (int k = 0; k < 8; k++) {
            int kl = (k < 4) ? (2 * k + 8 * khalf) : (16 + 2 * (k - 4) + 8 * khalf);
            a00[2 * k]     = sH[w][hh][kl];
            a00[2 * k + 1] = sH[w][hh][kl + 1];
            a01[2 * k]     = sH[w][hh][32 + kl];
            a01[2 * k + 1] = sH[w][hh][32 + kl + 1];
            a10[2 * k]     = sH[w][16 + hh][kl];
            a10[2 * k + 1] = sH[w][16 + hh][kl + 1];
            a11[2 * k]     = sH[w][16 + hh][32 + kl];
            a11[2 * k + 1] = sH[w][16 + hh][32 + kl + 1];
        }
        // ---- hoist the 16 feature values this lane needs (C/D row M = r + 8*khalf)
        float ff0[8], ff1[8];
        #pragma unroll
        for (int r = 0; r < 8; r++) {
            ff0[r] = sFeat[w][8 * khalf + r][i];
            ff1[r] = sFeat[w][16 + 8 * khalf + r][i];
        }

        float accO[8];
        #pragma unroll
        for (int t = 0; t < 8; t++) {             // 8 column tiles cover 128 att channels
            int n = t * 16 + hh;                  // global att column for this lane
            v16h b0, b1;
            #pragma unroll
            for (int k = 0; k < 8; k++) {
                int kl = (k < 4) ? (2 * k + 8 * khalf) : (16 + 2 * (k - 4) + 8 * khalf);
                b0[2 * k]     = sW2[n * 64 + kl];
                b0[2 * k + 1] = sW2[n * 64 + kl + 1];
                b1[2 * k]     = sW2[n * 64 + 32 + kl];
                b1[2 * k + 1] = sW2[n * 64 + 32 + kl + 1];
            }
            v8f c0 = {}, c1 = {};
            c0 = __builtin_amdgcn_wmma_f32_16x16x32_f16(false, a00, false, b0, (short)0, c0, false, false);
            c0 = __builtin_amdgcn_wmma_f32_16x16x32_f16(false, a01, false, b1, (short)0, c0, false, false);
            c1 = __builtin_amdgcn_wmma_f32_16x16x32_f16(false, a10, false, b0, (short)0, c1, false, false);
            c1 = __builtin_amdgcn_wmma_f32_16x16x32_f16(false, a11, false, b1, (short)0, c1, false, false);
            // consume att tile in-register: partial for output o = 2t + (hh>>3)
            float partial = 0.0f;
            #pragma unroll
            for (int r = 0; r < 8; r++)
                partial = fmaf(c0[r], ff0[r], fmaf(c1[r], ff1[r], partial));
            accO[t] = partial;
        }
        // one batch of LDS float atomics; wave-level DS ordering guarantees visibility
        #pragma unroll
        for (int t = 0; t < 8; t++)
            atomicAdd(&sOut[w][2 * t + (hh >> 3)], accO[t]);

        if (lane < OUT_CHN) pooled[m * OUT_CHN + lane] = sOut[w][lane];
    }
}

// ---- BN2 statistics over pooled (M x 16)
__global__ __launch_bounds__(256)
void stats2_kernel(const float* __restrict__ pooled, float* __restrict__ ws, int M) {
    __shared__ float s[256], s2[256];
    int tid = threadIdx.x, ch = tid & 15;
    float a = 0.0f, b = 0.0f;
    for (int row = blockIdx.x * 16 + (tid >> 4); row < M; row += gridDim.x * 16) {
        float p = pooled[row * OUT_CHN + ch];
        a += p; b += p * p;
    }
    s[tid] = a; s2[tid] = b;
    __syncthreads();
    for (int off = 128; off >= 16; off >>= 1) {
        if (tid < off) { s[tid] += s[tid + off]; s2[tid] += s2[tid + off]; }
        __syncthreads();
    }
    if (tid < 16) { atomicAdd(&ws[168 + tid], s[tid]); atomicAdd(&ws[184 + tid], s2[tid]); }
}

// ---- finalize: out = relu(BN2(pooled)); copy new_xyz into output slot 0
__global__ __launch_bounds__(256)
void finalize_kernel(const float* __restrict__ new_xyz, const float* __restrict__ pooled,
                     const float* __restrict__ ws, const float* __restrict__ gamma2,
                     const float* __restrict__ beta2, float* __restrict__ out, int M) {
    int idx = blockIdx.x * blockDim.x + threadIdx.x;
    int nxyz = M * 3;
    if (idx < nxyz) out[idx] = new_xyz[idx];
    int j = idx - nxyz;
    if (j >= 0 && j < M * OUT_CHN) {
        int c = j & 15;
        float invM = 1.0f / (float)M;
        float mu  = ws[168 + c] * invM;
        float var = ws[184 + c] * invM - mu * mu;
        float rs  = rsqrtf(fmaxf(var, 0.0f) + EPS);
        float v   = (pooled[j] - mu) * rs * gamma2[c] + beta2[c];
        out[nxyz + j] = fmaxf(v, 0.0f);
    }
}

extern "C" void kernel_launch(void* const* d_in, const int* in_sizes, int n_in,
                              void* d_out, int out_size, void* d_ws, size_t ws_size,
                              hipStream_t stream) {
    const float* xyz      = (const float*)d_in[0];
    const float* new_xyz  = (const float*)d_in[1];
    const float* features = (const float*)d_in[2];
    const int*   group_idx= (const int*)d_in[3];
    const float* W1       = (const float*)d_in[4];
    const float* g1       = (const float*)d_in[5];
    const float* b1       = (const float*)d_in[6];
    const float* W2       = (const float*)d_in[7];
    const float* g2       = (const float*)d_in[8];
    const float* b2       = (const float*)d_in[9];
    float* out = (float*)d_out;
    float* ws  = (float*)d_ws;

    int M = in_sizes[1] / 3;        // 8192
    int L = in_sizes[3];            // M * NS
    float* pooled = ws + 256;

    int zn = 256 + M * OUT_CHN;
    zero_kernel<<<(zn + 255) / 256, 256, 0, stream>>>(ws, zn);
    stats1_kernel<<<(M + WPB - 1) / WPB, 256, 0, stream>>>(xyz, new_xyz, group_idx, ws, M);
    bn1_finalize_kernel<<<1, 64, 0, stream>>>(W1, g1, b1, ws, 1.0f / (float)L);
    attn_wmma_kernel<<<(M + WPB - 1) / WPB, 256, 0, stream>>>(
        xyz, new_xyz, features, group_idx, W1, W2, ws + 40, ws + 104, pooled, M);
    stats2_kernel<<<32, 256, 0, stream>>>(pooled, ws, M);
    int fin = M * 19;               // M*3 + M*16
    finalize_kernel<<<(fin + 255) / 256, 256, 0, stream>>>(new_xyz, pooled, ws, g2, b2, out, M);
}
